// PortfolioEncoder_50972671869495
// MI455X (gfx1250) — matrix operand order repaired and checked by reading, not compiled, not run on hardware
//
#include <hip/hip_runtime.h>
#include <hip/hip_bf16.h>

// ---------------------------------------------------------------------------
// PortfolioEncoder for MI455X (gfx1250, wave32, WMMA + async LDS copies).
// B=32, T=2048, D=1024, H=16, HD=64.
// Dominant cost: (B*T x D) x (D x 2D) K/V projection GEMM (~275 GFLOP)
//  -> stage LN(H_p) and W as bf16 once (HBM-cheap: ~20us at 23.3 TB/s),
//     then a pure bf16 WMMA GEMM with double-buffered
//     GLOBAL_LOAD_ASYNC_TO_LDS_B128 tile copies (ASYNCcnt pipeline).
// ---------------------------------------------------------------------------

#define BB 32
#define TT 2048
#define DD 1024
#define HH 16
#define HD 64
#define MROWS (BB * TT)   // 65536
#define EPS 1e-5f

typedef __bf16 v16bf __attribute__((ext_vector_type(16)));
typedef float  v8f   __attribute__((ext_vector_type(8)));
typedef int    v4i   __attribute__((vector_size(16)));

#define GAS __attribute__((address_space(1)))
#define LAS __attribute__((address_space(3)))

#if __has_builtin(__builtin_amdgcn_global_load_async_to_lds_b128) && \
    __has_builtin(__builtin_amdgcn_s_wait_asynccnt)
#define USE_ASYNC_LDS 1
#else
#define USE_ASYNC_LDS 0
#endif

static __device__ __forceinline__ unsigned short f2bf(float f) {
    unsigned int u = __float_as_uint(f);
    u += 0x7fffu + ((u >> 16) & 1u);   // round-to-nearest-even
    return (unsigned short)(u >> 16);
}
static __device__ __forceinline__ float bf2f(unsigned short h) {
    return __uint_as_float(((unsigned int)h) << 16);
}
static __device__ __forceinline__ unsigned int pack2(float a, float b) {
    return (unsigned int)f2bf(a) | ((unsigned int)f2bf(b) << 16);
}

// ---------------------------------------------------------------------------
// Kernel 1: kv_bf16[row][k] = bf16( LN(H_p[row]) )  (65536 rows of 1024).
// One block (256 threads) per row.
// ---------------------------------------------------------------------------
__global__ __launch_bounds__(256) void ln_norm_kernel(
    const float* __restrict__ X, const float* __restrict__ g,
    const float* __restrict__ bvec, unsigned short* __restrict__ Y) {
    int row = blockIdx.x;
    int tid = threadIdx.x;
    const float* x = X + (size_t)row * DD;
    float4 v = *(const float4*)(x + tid * 4);
    __shared__ float2 red[256];
    red[tid] = make_float2(v.x + v.y + v.z + v.w,
                           v.x * v.x + v.y * v.y + v.z * v.z + v.w * v.w);
    __syncthreads();
    for (int o = 128; o > 0; o >>= 1) {
        if (tid < o) {
            red[tid].x += red[tid + o].x;
            red[tid].y += red[tid + o].y;
        }
        __syncthreads();
    }
    float mu   = red[0].x * (1.0f / DD);
    float var  = red[0].y * (1.0f / DD) - mu * mu;
    float rstd = rsqrtf(var + EPS);
    int k = tid * 4;
    float o0 = (v.x - mu) * rstd * g[k + 0] + bvec[k + 0];
    float o1 = (v.y - mu) * rstd * g[k + 1] + bvec[k + 1];
    float o2 = (v.z - mu) * rstd * g[k + 2] + bvec[k + 2];
    float o3 = (v.w - mu) * rstd * g[k + 3] + bvec[k + 3];
    uint2 p = make_uint2(pack2(o0, o1), pack2(o2, o3));
    *(uint2*)(Y + (size_t)row * DD + k) = p;
}

// ---------------------------------------------------------------------------
// Kernel 2: bf16 conversion of W (rows D..3D of in_proj_w), 2048x1024.
// ---------------------------------------------------------------------------
__global__ __launch_bounds__(256) void w_convert_kernel(
    const float* __restrict__ W, unsigned short* __restrict__ Wb) {
    size_t i = ((size_t)blockIdx.x * 256 + threadIdx.x) * 8;
    float4 a = *(const float4*)(W + i);
    float4 b = *(const float4*)(W + i + 4);
    uint4 p = make_uint4(pack2(a.x, a.y), pack2(a.z, a.w),
                         pack2(b.x, b.y), pack2(b.z, b.w));
    *(uint4*)(Wb + i) = p;
}

// ---------------------------------------------------------------------------
// Kernel 3: q = LN(h_t)*tau; qh = q @ Wq^T + bq.  One block per batch row.
// ---------------------------------------------------------------------------
__global__ __launch_bounds__(256) void q_proj_kernel(
    const float* __restrict__ h_t,
    const float* __restrict__ ln_q_g, const float* __restrict__ ln_q_b,
    const float* __restrict__ Wq,     const float* __restrict__ bq,
    const float* __restrict__ log_tau, float* __restrict__ qh) {
    int b = blockIdx.x;
    int tid = threadIdx.x;
    __shared__ float  q[DD];
    __shared__ float2 red[256];
    const float* x = h_t + (size_t)b * DD;
    float4 v = *(const float4*)(x + tid * 4);
    red[tid] = make_float2(v.x + v.y + v.z + v.w,
                           v.x * v.x + v.y * v.y + v.z * v.z + v.w * v.w);
    __syncthreads();
    for (int o = 128; o > 0; o >>= 1) {
        if (tid < o) {
            red[tid].x += red[tid + o].x;
            red[tid].y += red[tid + o].y;
        }
        __syncthreads();
    }
    float mu   = red[0].x * (1.0f / DD);
    float var  = red[0].y * (1.0f / DD) - mu * mu;
    float rstd = rsqrtf(var + EPS);
    float tau  = __expf(log_tau[0]);
    tau = fminf(fmaxf(tau, 0.25f), 4.0f);
    #pragma unroll
    for (int i = 0; i < 4; i++) {
        int k = tid * 4 + i;
        float xv = (i == 0) ? v.x : (i == 1) ? v.y : (i == 2) ? v.z : v.w;
        q[k] = ((xv - mu) * rstd * ln_q_g[k] + ln_q_b[k]) * tau;
    }
    __syncthreads();
    #pragma unroll
    for (int c = 0; c < 4; c++) {
        int col = tid + c * 256;
        const float* wrow = Wq + (size_t)col * DD;
        float acc = bq[col];
        for (int k = 0; k < DD; k++) acc += q[k] * wrow[k];
        qh[(size_t)b * DD + col] = acc;
    }
}

// ---------------------------------------------------------------------------
// Kernel 4: pure bf16 WMMA GEMM, C[65536, 2048] = A @ W^T + bias -> bf16.
//   Block tile 128x128, K-chunk 32, double-buffered LDS fed by
//   GLOBAL_LOAD_ASYNC_TO_LDS_B128 (fallback: sync b128 + ds_store).
//   8 waves: wave (wm 0..3, wn 0..1) computes rows [wm*32,+32) x cols
//   [wn*64,+64) = 8 WMMA 16x16 tiles (64 accumulator VGPRs).
// NOTE: one extra prefetch stage is issued past the K loop so the asynccnt
//   wait stays a constant immediate; its 64B over-read lands in the adjacent
//   workspace region (kv_bf16 -> w_bf16 -> kvh), which is safe.
// ---------------------------------------------------------------------------
#define BM 128
#define BN 128
#define BK 32
#define KSTEPS (DD / BK)   // 32

__global__ __launch_bounds__(256) void kv_proj_gemm_kernel(
    const unsigned short* __restrict__ A,   // kv_bf16 [65536][1024]
    const unsigned short* __restrict__ Bw,  // w_bf16  [2048][1024]
    const float* __restrict__ bias,         // in_proj_b + D
    unsigned short* __restrict__ kvh) {
    __shared__ __align__(128) unsigned short As[2][BM * BK];  // 2 x 8 KB
    __shared__ __align__(128) unsigned short Bs[2][BN * BK];  // 2 x 8 KB

    int tid  = threadIdx.x;
    int n0   = blockIdx.x * BN;     // n fastest -> A row-tile reuse in L2
    int m0   = blockIdx.y * BM;
    int lane = tid & 31;
    int wave = tid >> 5;
    int wm   = wave & 3;            // rows [wm*32, +32)
    int wn   = wave >> 2;           // cols [wn*64, +64)

    // Tile copy: 128 rows x 64B = 512 chunks of 16B; 2 chunks/thread/operand.
    int c0 = tid, c1 = tid + 256;
    const unsigned short* gA0 = A  + (size_t)(m0 + (c0 >> 2)) * DD + (c0 & 3) * 8;
    const unsigned short* gA1 = A  + (size_t)(m0 + (c1 >> 2)) * DD + (c1 & 3) * 8;
    const unsigned short* gB0 = Bw + (size_t)(n0 + (c0 >> 2)) * DD + (c0 & 3) * 8;
    const unsigned short* gB1 = Bw + (size_t)(n0 + (c1 >> 2)) * DD + (c1 & 3) * 8;

    v8f acc[2][4] = {};

    auto issue = [&](int buf, int ks) {
        int ofs = ks * BK;  // elements along K
#if USE_ASYNC_LDS
        __builtin_amdgcn_global_load_async_to_lds_b128(
            (GAS v4i*)(gA0 + ofs), (LAS v4i*)&As[buf][c0 * 8], 0, 0);
        __builtin_amdgcn_global_load_async_to_lds_b128(
            (GAS v4i*)(gA1 + ofs), (LAS v4i*)&As[buf][c1 * 8], 0, 0);
        __builtin_amdgcn_global_load_async_to_lds_b128(
            (GAS v4i*)(gB0 + ofs), (LAS v4i*)&Bs[buf][c0 * 8], 0, 0);
        __builtin_amdgcn_global_load_async_to_lds_b128(
            (GAS v4i*)(gB1 + ofs), (LAS v4i*)&Bs[buf][c1 * 8], 0, 0);
#else
        uint4 ta0 = *(const uint4*)(gA0 + ofs);
        uint4 ta1 = *(const uint4*)(gA1 + ofs);
        uint4 tb0 = *(const uint4*)(gB0 + ofs);
        uint4 tb1 = *(const uint4*)(gB1 + ofs);
        *(uint4*)&As[buf][c0 * 8] = ta0;
        *(uint4*)&As[buf][c1 * 8] = ta1;
        *(uint4*)&Bs[buf][c0 * 8] = tb0;
        *(uint4*)&Bs[buf][c1 * 8] = tb1;
#endif
    };

    issue(0, 0);
    for (int ks = 0; ks < KSTEPS; ks++) {
        int p = ks & 1;
        issue(p ^ 1, ks + 1);   // stage ks+1 (one harmless extra at the end)
#if USE_ASYNC_LDS
        __builtin_amdgcn_s_wait_asynccnt(4);  // retire stage ks, keep ks+1 inflight
#endif
        __syncthreads();        // all waves' stage-ks tiles visible

        #pragma unroll
        for (int half = 0; half < 2; half++) {
            // A fragment 16x32 bf16 in contiguous K order, same as B:
            // lanes 0-15 : row=lane,    K = {0..15}
            // lanes 16-31: row=lane-16, K = {16..31}
            v16bf afrag;
            {
                union { v16bf v; unsigned short u[16]; } ua;
                int r  = wm * 32 + half * 16 + (lane & 15);
                int kb = (lane < 16) ? 0 : 16;
                #pragma unroll
                for (int i = 0; i < 8; i++) ua.u[i]     = As[p][r * BK + kb + i];
                #pragma unroll
                for (int i = 0; i < 8; i++) ua.u[8 + i] = As[p][r * BK + kb + 8 + i];
                afrag = ua.v;
            }
            #pragma unroll
            for (int s = 0; s < 4; s++) {
                // B fragment: col = lane&15, K half by lane>=16; Bs is
                // N-major so the 16 K-values are 32 contiguous bytes.
                int col = wn * 64 + s * 16 + (lane & 15);
                int kb  = (lane < 16) ? 0 : 16;
                v16bf bfrag = *(const v16bf*)&Bs[p][col * BK + kb];
                acc[half][s] = __builtin_amdgcn_wmma_f32_16x16x32_bf16(
                    false, afrag, false, bfrag, (short)0, acc[half][s],
                    false, false);
            }
        }
        __syncthreads();        // done reading buffer p before it is re-filled
    }

    // D layout: VGPR j -> (M=j, N=lane) lanes 0-15 ; (M=j+8, N=lane-16)
    #pragma unroll
    for (int half = 0; half < 2; half++) {
        #pragma unroll
        for (int s = 0; s < 4; s++) {
            int gcol = n0 + wn * 64 + s * 16 + (lane & 15);
            float bb = bias[gcol];
            #pragma unroll
            for (int j = 0; j < 8; j++) {
                int grow = m0 + wm * 32 + half * 16 + j + ((lane >= 16) ? 8 : 0);
                kvh[(size_t)grow * (2 * DD) + gcol] = f2bf(acc[half][s][j] + bb);
            }
        }
    }
}

// ---------------------------------------------------------------------------
// Kernel 5: attention per (b,h): scores -> masked softmax -> ctx.
// kvh row layout: cols [0,1024) = K-proj, [1024,2048) = V-proj.
// ---------------------------------------------------------------------------
__global__ __launch_bounds__(256) void attn_kernel(
    const float* __restrict__ qh, const unsigned short* __restrict__ kvh,
    const int* __restrict__ valid_len, float* __restrict__ ctx) {
    int bh = blockIdx.x;
    int b = bh >> 4, h = bh & 15;
    int tid = threadIdx.x;
    __shared__ float sc[TT];
    __shared__ float qs[HD];
    __shared__ float red[256];

    if (tid < HD) qs[tid] = qh[(size_t)b * DD + h * HD + tid];
    __syncthreads();
    int vlen = valid_len[b];

    for (int t = tid; t < TT; t += 256) {
        const unsigned int* kp =
            (const unsigned int*)(kvh + ((size_t)(b * TT + t)) * (2 * DD) + h * HD);
        float acc = 0.f;
        #pragma unroll
        for (int i = 0; i < 32; i++) {
            unsigned int p = kp[i];
            acc += bf2f((unsigned short)(p & 0xffffu)) * qs[2 * i];
            acc += bf2f((unsigned short)(p >> 16))     * qs[2 * i + 1];
        }
        sc[t] = (t < vlen) ? acc * 0.125f : -3.0e38f;
    }
    __syncthreads();

    float lm = -3.0e38f;
    for (int t = tid; t < TT; t += 256) lm = fmaxf(lm, sc[t]);
    red[tid] = lm;
    __syncthreads();
    for (int o = 128; o > 0; o >>= 1) {
        if (tid < o) red[tid] = fmaxf(red[tid], red[tid + o]);
        __syncthreads();
    }
    float mx = red[0];
    __syncthreads();

    float ls = 0.f;
    for (int t = tid; t < TT; t += 256) {
        float w = __expf(sc[t] - mx);
        sc[t] = w;
        ls += w;
    }
    red[tid] = ls;
    __syncthreads();
    for (int o = 128; o > 0; o >>= 1) {
        if (tid < o) red[tid] += red[tid + o];
        __syncthreads();
    }
    float inv = 1.0f / red[0];
    __syncthreads();

    int d = tid & 63, tq = tid >> 6;
    float acc = 0.f;
    for (int t = tq; t < TT; t += 4) {
        acc += sc[t] *
               bf2f(kvh[((size_t)(b * TT + t)) * (2 * DD) + DD + h * HD + d]);
    }
    red[tid] = acc;
    __syncthreads();
    if (tid < HD) {
        float r = red[tid] + red[tid + 64] + red[tid + 128] + red[tid + 192];
        ctx[(size_t)b * DD + h * HD + tid] = r * inv;
    }
}

// ---------------------------------------------------------------------------
// Kernel 6: out = LN(ctx @ out_w^T + out_b + h_t).  One block per batch row.
// ---------------------------------------------------------------------------
__global__ __launch_bounds__(256) void out_proj_kernel(
    const float* __restrict__ ctx, const float* __restrict__ h_t,
    const float* __restrict__ W,   const float* __restrict__ bias,
    const float* __restrict__ g,   const float* __restrict__ bvec,
    float* __restrict__ out) {
    int b = blockIdx.x;
    int tid = threadIdx.x;
    __shared__ float  cs[DD];
    __shared__ float  y[DD];
    __shared__ float2 red[256];
    *(float4*)(cs + tid * 4) = *(const float4*)(ctx + (size_t)b * DD + tid * 4);
    __syncthreads();
    #pragma unroll
    for (int c = 0; c < 4; c++) {
        int col = tid + c * 256;
        const float* wrow = W + (size_t)col * DD;
        float acc = bias[col];
        for (int k = 0; k < DD; k++) acc += cs[k] * wrow[k];
        y[col] = acc + h_t[(size_t)b * DD + col];
    }
    __syncthreads();
    float s = 0.f, s2 = 0.f;
    #pragma unroll
    for (int i = 0; i < 4; i++) {
        float v = y[tid * 4 + i];
        s += v; s2 += v * v;
    }
    red[tid] = make_float2(s, s2);
    __syncthreads();
    for (int o = 128; o > 0; o >>= 1) {
        if (tid < o) {
            red[tid].x += red[tid + o].x;
            red[tid].y += red[tid + o].y;
        }
        __syncthreads();
    }
    float mu   = red[0].x * (1.0f / DD);
    float var  = red[0].y * (1.0f / DD) - mu * mu;
    float rstd = rsqrtf(var + EPS);
    #pragma unroll
    for (int i = 0; i < 4; i++) {
        int k = tid * 4 + i;
        out[(size_t)b * DD + k] = (y[k] - mu) * rstd * g[k] + bvec[k];
    }
}

// ---------------------------------------------------------------------------
// Launch. Workspace layout (needs ~389 MB):
//   [0, 128K)                      float qh[32*1024]
//   [128K, 256K)                   float ctx[32*1024]
//   [1M, 1M+128M)                  bf16  kv_bf16[65536*1024]
//   [1M+128M, 1M+132M)             bf16  w_bf16[2048*1024]
//   [1M+132M, 1M+132M+256M)        bf16  kvh[65536*2048]
// ---------------------------------------------------------------------------
extern "C" void kernel_launch(void* const* d_in, const int* in_sizes, int n_in,
                              void* d_out, int out_size, void* d_ws, size_t ws_size,
                              hipStream_t stream) {
    const float* h_t       = (const float*)d_in[0];
    const float* H_p       = (const float*)d_in[1];
    const int*   valid_len = (const int*)d_in[2];
    const float* ln_q_g    = (const float*)d_in[3];
    const float* ln_q_b    = (const float*)d_in[4];
    const float* ln_kv_g   = (const float*)d_in[5];
    const float* ln_kv_b   = (const float*)d_in[6];
    const float* ln_out_g  = (const float*)d_in[7];
    const float* ln_out_b  = (const float*)d_in[8];
    const float* log_tau   = (const float*)d_in[9];
    const float* in_proj_w = (const float*)d_in[10];
    const float* in_proj_b = (const float*)d_in[11];
    const float* out_w     = (const float*)d_in[12];
    const float* out_b     = (const float*)d_in[13];
    float* out = (float*)d_out;

    char* ws = (char*)d_ws;
    float*          qh      = (float*)ws;
    float*          ctx     = (float*)(ws + (size_t)(128 << 10));
    const size_t    MB      = (size_t)1 << 20;
    unsigned short* kv_bf16 = (unsigned short*)(ws + MB);
    unsigned short* w_bf16  = (unsigned short*)(ws + MB + 128 * MB);
    unsigned short* kvh     = (unsigned short*)(ws + MB + 132 * MB);

    ln_norm_kernel<<<MROWS, 256, 0, stream>>>(H_p, ln_kv_g, ln_kv_b, kv_bf16);

    // rows D..3D of in_proj_w -> bf16 (2048*1024 elems, 8/thread)
    w_convert_kernel<<<(2 * DD * DD) / (256 * 8), 256, 0, stream>>>(
        in_proj_w + (size_t)DD * DD, w_bf16);

    q_proj_kernel<<<BB, 256, 0, stream>>>(
        h_t, ln_q_g, ln_q_b, in_proj_w, in_proj_b, log_tau, qh);

    dim3 ggrid(2 * DD / BN, MROWS / BM);   // (16, 512): n fastest for L2 reuse
    kv_proj_gemm_kernel<<<ggrid, 256, 0, stream>>>(
        kv_bf16, w_bf16, in_proj_b + DD, kvh);

    attn_kernel<<<BB * HH, 256, 0, stream>>>(qh, kvh, valid_len, ctx);

    out_proj_kernel<<<BB, 256, 0, stream>>>(
        ctx, h_t, out_w, out_b, ln_out_g, ln_out_b, out);
}
